// MultiHeadAttention_18915035972266
// MI455X (gfx1250) — matrix operand order repaired and checked by reading, not compiled, not run on hardware
//
#include <hip/hip_runtime.h>

typedef __attribute__((ext_vector_type(16))) __bf16 v16bf;
typedef __attribute__((ext_vector_type(8)))  float  v8f;
typedef unsigned short ushort_t;

#define D_MODEL 1024
#define D_V     64
#define S_LEN   1024
#define BATCH   4
#define ROWS    (BATCH * S_LEN)        // 4096
#define HB      64                     // bs*heads head-batches
#define AT_ELEMS (HB * S_LEN * S_LEN)  // 67108864

// ---------- fp32 -> bf16 (RNE) ----------
__device__ __forceinline__ ushort_t f2bf(float f) {
    union { float f; unsigned u; } v; v.f = f;
    unsigned u = v.u;
    unsigned r = u + 0x7FFFu + ((u >> 16) & 1u);
    return (ushort_t)(r >> 16);
}

// ---------- WMMA fragment loaders (bf16 16x16x32, ISA 7.12.2 layout) ----------
// A (16x32, MxK): row = lane%16; lanes 0-15 hold K {0..7,16..23}, lanes 16-31 hold K {8..15,24..31}
__device__ __forceinline__ v16bf load_frag_a(const ushort_t* base, int ld, int row, int k0, int lane) {
    int r = row + (lane & 15);
    int ksel = (lane < 16) ? 0 : 8;
    const ushort_t* p = base + (size_t)r * ld + k0 + ksel;
    union { uint4 q[2]; v16bf v; } u;
    u.q[0] = *(const uint4*)p;          // K = k0+ksel .. +7
    u.q[1] = *(const uint4*)(p + 16);   // K = k0+16+ksel .. +7
    return u.v;
}
// B (32x16, KxN) loaded from TRANSPOSED storage BT[N,K]: col = lane%16;
// lanes 0-15 hold K {0..7,16..23}, lanes 16-31 hold K {8..15,24..31}
__device__ __forceinline__ v16bf load_frag_bt(const ushort_t* baseT, int ld, int col, int k0, int lane) {
    int c = col + (lane & 15);
    int kk = k0 + ((lane < 16) ? 0 : 8);
    const ushort_t* p = baseT + (size_t)c * ld + kk;
    union { uint4 q[2]; v16bf v; } u;
    u.q[0] = *(const uint4*)p;
    u.q[1] = *(const uint4*)(p + 16);
    return u.v;
}

__device__ __forceinline__ v8f wmma_bf16(v16bf a, v16bf b, v8f c) {
    return __builtin_amdgcn_wmma_f32_16x16x32_bf16(false, a, false, b, (short)0, c, false, false);
}

// ---------- convert kernels ----------
__global__ __launch_bounds__(256) void cvt_bf16_kernel(const float* __restrict__ src,
                                                       ushort_t* __restrict__ dst, int n) {
    int i = blockIdx.x * 256 + threadIdx.x;
    int stride = gridDim.x * 256;
    for (; i < n; i += stride) dst[i] = f2bf(src[i]);
}

// src [K,N] f32 row-major  ->  dst [N,K] bf16 row-major (transpose)
__global__ __launch_bounds__(256) void cvtT_bf16_kernel(const float* __restrict__ src,
                                                        ushort_t* __restrict__ dst, int K, int N) {
    int i = blockIdx.x * 256 + threadIdx.x;
    if (i >= K * N) return;
    int k = i / N, n = i - k * N;
    dst[(size_t)n * K + k] = f2bf(src[i]);
}

// ---------- generic bf16 WMMA GEMM: C[M,N] = A[M,K] * BT[N,K]^T + bias (+resid) ----------
__global__ __launch_bounds__(256) void gemm_bf16_kernel(
    const ushort_t* __restrict__ A, const ushort_t* __restrict__ BT,
    const float* __restrict__ bias, const float* __restrict__ resid,
    float* __restrict__ Cf, ushort_t* __restrict__ Cb, int M, int N, int K) {
    int lane = threadIdx.x & 31, wid = threadIdx.x >> 5;
    int row_base = blockIdx.y * 128 + (wid & 3) * 32;
    int col_base = blockIdx.x * 128 + (wid >> 2) * 64;

    v8f zero = {0.f, 0.f, 0.f, 0.f, 0.f, 0.f, 0.f, 0.f};
    v8f acc[2][4];
#pragma unroll
    for (int i = 0; i < 2; i++)
#pragma unroll
        for (int j = 0; j < 4; j++) acc[i][j] = zero;

    for (int k0 = 0; k0 < K; k0 += 32) {
        v16bf a0 = load_frag_a(A, K, row_base, k0, lane);
        v16bf a1 = load_frag_a(A, K, row_base + 16, k0, lane);
#pragma unroll
        for (int j = 0; j < 4; j++) {
            v16bf b = load_frag_bt(BT, K, col_base + j * 16, k0, lane);
            acc[0][j] = wmma_bf16(a0, b, acc[0][j]);
            acc[1][j] = wmma_bf16(a1, b, acc[1][j]);
        }
    }

    int colb = lane & 15;
    int rsel = (lane < 16) ? 0 : 8;
#pragma unroll
    for (int i = 0; i < 2; i++)
#pragma unroll
        for (int j = 0; j < 4; j++) {
            int col = col_base + j * 16 + colb;
            float bv = bias ? bias[col] : 0.f;
#pragma unroll
            for (int r = 0; r < 8; r++) {
                int row = row_base + i * 16 + r + rsel;
                size_t idx = (size_t)row * N + col;
                float v = acc[i][j][r] + bv;
                if (resid) v += resid[idx];
                if (Cf) Cf[idx] = v;
                if (Cb) Cb[idx] = f2bf(v);
            }
        }
}

// ---------- scores: at[z] = Q_z (1024x64) * K_z^T / 8, per head-batch z ----------
__global__ __launch_bounds__(256) void scores_kernel(const ushort_t* __restrict__ qp,
                                                     const ushort_t* __restrict__ kp,
                                                     float* __restrict__ at) {
    int z = blockIdx.z;
    const ushort_t* Q  = qp + (size_t)z * 65536;   // [1024,64] bf16
    const ushort_t* Kh = kp + (size_t)z * 65536;   // [1024,64] bf16 (acts as B^T storage)
    int lane = threadIdx.x & 31, wid = threadIdx.x >> 5;
    int row_base = blockIdx.y * 128 + (wid & 3) * 32;
    int col_base = blockIdx.x * 128 + (wid >> 2) * 64;

    v8f zero = {0.f, 0.f, 0.f, 0.f, 0.f, 0.f, 0.f, 0.f};
    v8f acc[2][4];
#pragma unroll
    for (int i = 0; i < 2; i++)
#pragma unroll
        for (int j = 0; j < 4; j++) acc[i][j] = zero;

#pragma unroll
    for (int k0 = 0; k0 < 64; k0 += 32) {
        v16bf a0 = load_frag_a(Q, 64, row_base, k0, lane);
        v16bf a1 = load_frag_a(Q, 64, row_base + 16, k0, lane);
#pragma unroll
        for (int j = 0; j < 4; j++) {
            v16bf b = load_frag_bt(Kh, 64, col_base + j * 16, k0, lane);
            acc[0][j] = wmma_bf16(a0, b, acc[0][j]);
            acc[1][j] = wmma_bf16(a1, b, acc[1][j]);
        }
    }

    float* atz = at + (size_t)z * 1048576;
    int colb = lane & 15;
    int rsel = (lane < 16) ? 0 : 8;
#pragma unroll
    for (int i = 0; i < 2; i++)
#pragma unroll
        for (int j = 0; j < 4; j++) {
            int col = col_base + j * 16 + colb;
#pragma unroll
            for (int r = 0; r < 8; r++) {
                int row = row_base + i * 16 + r + rsel;
                atz[(size_t)row * 1024 + col] = acc[i][j][r] * 0.125f;
            }
        }
}

// ---------- fused softmax + attn*V per 64-row block of one head-batch ----------
__global__ __launch_bounds__(256) void softmax_av_kernel(const float* __restrict__ at,
                                                         const ushort_t* __restrict__ vp,
                                                         ushort_t* __restrict__ ob) {
    __shared__ float sm[64];           // row max
    __shared__ float sr[64];           // 1/rowsum
    __shared__ ushort_t pS[64][64];    // probs chunk (bf16)
    __shared__ ushort_t vT[64][64];    // V chunk transposed: vT[n][k]
    int z = blockIdx.y;
    int row0 = blockIdx.x * 64;
    int tid = threadIdx.x, lane = tid & 31, wid = tid >> 5;
    const float* atz = at + (size_t)z * 1048576;
    const ushort_t* vz = vp + (size_t)z * 65536;

    // Stage 1: online softmax stats (8 rows per wave, lane-strided over 1024 cols)
    for (int rr = 0; rr < 8; rr++) {
        int r = wid * 8 + rr;
        const float* rowp = atz + (size_t)(row0 + r) * 1024;
        float m = -3.0e38f, s = 0.f;
        for (int i = 0; i < 32; i++) {
            float x = rowp[lane + i * 32];
            float nm = fmaxf(m, x);
            s = s * __expf(m - nm) + __expf(x - nm);
            m = nm;
        }
        for (int off = 16; off > 0; off >>= 1) {
            float om = __shfl_xor(m, off, 32);
            float os = __shfl_xor(s, off, 32);
            float nm = fmaxf(m, om);
            s = s * __expf(m - nm) + os * __expf(om - nm);
            m = nm;
        }
        if (lane == 0) { sm[r] = m; sr[r] = 1.0f / s; }
    }
    __syncthreads();

    v8f zero = {0.f, 0.f, 0.f, 0.f, 0.f, 0.f, 0.f, 0.f};
    v8f acc[2];
    acc[0] = zero; acc[1] = zero;

    for (int kc = 0; kc < 16; kc++) {
        // probs chunk -> LDS (bf16); V chunk -> LDS transposed
        for (int idx = tid; idx < 4096; idx += 256) {
            int r = idx >> 6, c = idx & 63;
            float x = atz[(size_t)(row0 + r) * 1024 + kc * 64 + c];
            pS[r][c] = f2bf(__expf(x - sm[r]) * sr[r]);
            vT[c][r] = vz[(size_t)(kc * 64 + r) * 64 + c];
        }
        __syncthreads();
#pragma unroll
        for (int i = 0; i < 2; i++) {
            int t = wid * 2 + i, tm = t >> 2, tn = t & 3;
#pragma unroll
            for (int ks = 0; ks < 64; ks += 32) {
                v16bf a = load_frag_a(&pS[0][0], 64, tm * 16, ks, lane);
                v16bf b = load_frag_bt(&vT[0][0], 64, tn * 16, ks, lane);
                acc[i] = wmma_bf16(a, b, acc[i]);
            }
        }
        __syncthreads();
    }

    int colb = lane & 15;
    int rsel = (lane < 16) ? 0 : 8;
#pragma unroll
    for (int i = 0; i < 2; i++) {
        int t = wid * 2 + i, tm = t >> 2, tn = t & 3;
#pragma unroll
        for (int r = 0; r < 8; r++) {
            int row = row0 + tm * 16 + r + rsel;
            int col = tn * 16 + colb;
            ob[(size_t)z * 65536 + (size_t)row * 64 + col] = f2bf(acc[i][r]);
        }
    }
}

// ---------- LayerNorm over rows of x[4096,1024] ----------
__global__ __launch_bounds__(256) void ln_kernel(const float* __restrict__ x,
                                                 const float* __restrict__ gamma,
                                                 const float* __restrict__ beta,
                                                 float* __restrict__ out) {
    __shared__ float red[256];
    int row = blockIdx.x, tid = threadIdx.x;
    const float* xr = x + (size_t)row * 1024;
    float vals[4];
    float s = 0.f;
#pragma unroll
    for (int i = 0; i < 4; i++) { float v = xr[tid + i * 256]; vals[i] = v; s += v; }
    red[tid] = s; __syncthreads();
    for (int off = 128; off > 0; off >>= 1) {
        if (tid < off) red[tid] += red[tid + off];
        __syncthreads();
    }
    float mu = red[0] * (1.0f / 1024.0f);
    __syncthreads();
    float ss = 0.f;
#pragma unroll
    for (int i = 0; i < 4; i++) { float d = vals[i] - mu; ss += d * d; }
    red[tid] = ss; __syncthreads();
    for (int off = 128; off > 0; off >>= 1) {
        if (tid < off) red[tid] += red[tid + off];
        __syncthreads();
    }
    float rstd = rsqrtf(red[0] * (1.0f / 1024.0f) + 1e-5f);
#pragma unroll
    for (int i = 0; i < 4; i++) {
        int c = tid + i * 256;
        out[(size_t)row * 1024 + c] = (vals[i] - mu) * rstd * gamma[c] + beta[c];
    }
}

extern "C" void kernel_launch(void* const* d_in, const int* in_sizes, int n_in,
                              void* d_out, int out_size, void* d_ws, size_t ws_size,
                              hipStream_t stream) {
    const float* q     = (const float*)d_in[0];
    const float* k     = (const float*)d_in[1];
    const float* v     = (const float*)d_in[2];
    const float* wq    = (const float*)d_in[3];
    const float* bq    = (const float*)d_in[4];
    const float* wk    = (const float*)d_in[5];
    const float* bk    = (const float*)d_in[6];
    const float* wv    = (const float*)d_in[7];
    const float* bv    = (const float*)d_in[8];
    const float* wo    = (const float*)d_in[9];
    const float* bo    = (const float*)d_in[10];
    const float* gamma = (const float*)d_in[11];
    const float* beta  = (const float*)d_in[12];

    float* out    = (float*)d_out;
    float* at     = out;                 // [64,1024,1024] scaled scores
    float* ln_out = out + AT_ELEMS;      // [4,1024,1024]

    // workspace layout (bytes)
    const size_t MB = 1024 * 1024;
    char* ws = (char*)d_ws;
    ushort_t* QB  = (ushort_t*)(ws +  0 * MB);  // q bf16        [4096,1024]  8MB
    ushort_t* KB  = (ushort_t*)(ws +  8 * MB);  // k bf16                      8MB
    ushort_t* VB  = (ushort_t*)(ws + 16 * MB);  // v bf16                      8MB
    ushort_t* WQT = (ushort_t*)(ws + 24 * MB);  // wq^T bf16     [1024,1024]  2MB
    ushort_t* WKT = (ushort_t*)(ws + 26 * MB);
    ushort_t* WVT = (ushort_t*)(ws + 28 * MB);
    ushort_t* WOT = (ushort_t*)(ws + 30 * MB);
    ushort_t* QP  = (ushort_t*)(ws + 32 * MB);  // qp bf16       [4096,1024]  8MB
    ushort_t* KP  = (ushort_t*)(ws + 40 * MB);
    ushort_t* VP  = (ushort_t*)(ws + 48 * MB);
    ushort_t* OB  = (ushort_t*)(ws + 56 * MB);  // attn output bf16 (== op)   8MB
    float*    X   = (float*)   (ws + 64 * MB);  // pre-LN f32    [4096,1024] 16MB

    const int ACT_N = ROWS * D_MODEL;   // 4194304
    const int W_N   = D_MODEL * D_MODEL;

    // 1. convert activations and weights (weights transposed for B-fragment loads)
    cvt_bf16_kernel<<<2048, 256, 0, stream>>>(q, QB, ACT_N);
    cvt_bf16_kernel<<<2048, 256, 0, stream>>>(k, KB, ACT_N);
    cvt_bf16_kernel<<<2048, 256, 0, stream>>>(v, VB, ACT_N);
    cvtT_bf16_kernel<<<W_N / 256, 256, 0, stream>>>(wq, WQT, D_MODEL, D_MODEL);
    cvtT_bf16_kernel<<<W_N / 256, 256, 0, stream>>>(wk, WKT, D_MODEL, D_MODEL);
    cvtT_bf16_kernel<<<W_N / 256, 256, 0, stream>>>(wv, WVT, D_MODEL, D_MODEL);
    cvtT_bf16_kernel<<<W_N / 256, 256, 0, stream>>>(wo, WOT, D_MODEL, D_MODEL);

    // 2. Q/K/V projections (bias fused, bf16 out)
    dim3 gproj(D_MODEL / 128, ROWS / 128);   // (8, 32)
    gemm_bf16_kernel<<<gproj, 256, 0, stream>>>(QB, WQT, bq, nullptr, nullptr, QP,
                                                ROWS, D_MODEL, D_MODEL);
    gemm_bf16_kernel<<<gproj, 256, 0, stream>>>(KB, WKT, bk, nullptr, nullptr, KP,
                                                ROWS, D_MODEL, D_MODEL);
    gemm_bf16_kernel<<<gproj, 256, 0, stream>>>(VB, WVT, bv, nullptr, nullptr, VP,
                                                ROWS, D_MODEL, D_MODEL);

    // 3. scores at = Q K^T / sqrt(d_v)  -> d_out (first output)
    scores_kernel<<<dim3(8, 8, HB), 256, 0, stream>>>(QP, KP, at);

    // 4. softmax + attn*V (reads at from d_out)
    softmax_av_kernel<<<dim3(16, HB), 256, 0, stream>>>(at, VP, OB);

    // 5. output projection + bias + residual -> X (f32)
    gemm_bf16_kernel<<<gproj, 256, 0, stream>>>(OB, WOT, bo, q, X, nullptr,
                                                ROWS, D_MODEL, D_MODEL);

    // 6. LayerNorm -> second output
    ln_kernel<<<ROWS, 256, 0, stream>>>(X, gamma, beta, ln_out);
}
